// TransformerBlock_19731079758429
// MI455X (gfx1250) — matrix operand order repaired and checked by reading, not compiled, not run on hardware
//
#include <hip/hip_runtime.h>
#include <math.h>

// ---------------- types for CDNA5 WMMA ----------------
typedef __bf16 bf16_t;
typedef __attribute__((ext_vector_type(16))) __bf16 v16bf;
typedef __attribute__((ext_vector_type(8)))  __bf16 v8bf;
typedef __attribute__((ext_vector_type(8)))  float  v8f;

#define B_     2
#define S_     1024
#define H_     1024
#define NH_    16
#define HD_    64
#define E_     8
#define INNER_ 4096
#define T_     (B_ * S_)      // 2048 tokens
#define CAP_   320            // ceil(1.25 * 2048 / 8)
#define LDQ_   (3 * H_)       // qkv row stride

union Frag { v16bf v; v8bf h[2]; };

__device__ __forceinline__ v8f wmma_bf16(v16bf a, v16bf b, v8f c) {
  // v_wmma_f32_16x16x32_bf16  D = A(16x32) * B(32x16) + C
  return __builtin_amdgcn_wmma_f32_16x16x32_bf16(false, a, false, b, (short)0, c,
                                                 false, false);
}

// low 32 bits of a generic pointer to LDS == LDS byte offset (aperture model)
__device__ __forceinline__ unsigned lds_lo(const void* p) {
  return (unsigned)(unsigned long long)p;
}
// async global -> LDS, 16B per lane (ASYNCcnt)
__device__ __forceinline__ void async_b128(unsigned ldsAddr, const void* gaddr) {
  asm volatile("global_load_async_to_lds_b128 %0, %1, off"
               :: "v"(ldsAddr), "v"((unsigned long long)gaddr)
               : "memory");
}
__device__ __forceinline__ void wait_async0() {
  asm volatile("s_wait_asynccnt 0x0" ::: "memory");
}

// ---------------- weight convert + transpose: f32 [R,C] -> bf16 [C,R] ----------------
__global__ __launch_bounds__(256) void k_transpose_to_bf16(
    const float* __restrict__ in, bf16_t* __restrict__ out, int R, int C) {
  __shared__ float tile[32][33];
  const size_t batch = blockIdx.z;
  in  += batch * (size_t)R * C;
  out += batch * (size_t)R * C;
  const int c0 = blockIdx.x * 32, r0 = blockIdx.y * 32;
  const int tx = threadIdx.x, ty = threadIdx.y;   // 32 x 8
  for (int i = 0; i < 32; i += 8) {
    int r = r0 + ty + i, c = c0 + tx;
    tile[ty + i][tx] = (r < R && c < C) ? in[(size_t)r * C + c] : 0.f;
  }
  __syncthreads();
  for (int i = 0; i < 32; i += 8) {
    int c = c0 + ty + i, r = r0 + tx;
    if (c < C && r < R) out[(size_t)c * R + r] = (bf16_t)tile[tx][ty + i];
  }
}

// ---------------- RMSNorm (writes bf16, optional f32 copy) ----------------
__global__ __launch_bounds__(256) void k_rmsnorm(
    const float* __restrict__ x, const float* __restrict__ w,
    bf16_t* __restrict__ ob, float* __restrict__ of) {
  const int row = blockIdx.x;
  const float* xr = x + (size_t)row * H_;
  __shared__ float red[256];
  float s = 0.f;
  for (int c = threadIdx.x; c < H_; c += 256) { float v = xr[c]; s += v * v; }
  red[threadIdx.x] = s; __syncthreads();
  for (int o = 128; o > 0; o >>= 1) {
    if (threadIdx.x < o) red[threadIdx.x] += red[threadIdx.x + o];
    __syncthreads();
  }
  const float scale = rsqrtf(red[0] / (float)H_ + 1e-6f);
  for (int c = threadIdx.x; c < H_; c += 256) {
    float v = xr[c] * scale * w[c];
    ob[(size_t)row * H_ + c] = (bf16_t)v;
    if (of) of[(size_t)row * H_ + c] = v;
  }
}

// ---------------- block-tiled WMMA GEMM with async global->LDS double buffering ----------
// C[M,N] = A[M,K](bf16) * BT[N,K](bf16)^T (+bias)(+resid); block tile 64x128, BK=32,
// 8 waves each computing 32x32 (4 wmma / k-step) from LDS fragments.
#define GBM 64
#define GBN 128
#define GBK 32
__global__ __launch_bounds__(256) void k_gemm_wmma(
    const bf16_t* __restrict__ A, const bf16_t* __restrict__ BT,
    const float* __restrict__ bias, const float* __restrict__ resid,
    float* __restrict__ Cf, bf16_t* __restrict__ Cb,
    int M, int N, int K,
    size_t aStride, size_t bStride, size_t cStride, size_t biasStride) {
  __shared__ __align__(16) bf16_t Abuf[2][GBM * GBK];   // 2 x 4 KB
  __shared__ __align__(16) bf16_t Bbuf[2][GBN * GBK];   // 2 x 8 KB
  const int z = blockIdx.z;
  A  += (size_t)z * aStride;
  BT += (size_t)z * bStride;
  if (bias) bias += (size_t)z * biasStride;
  const int tid  = threadIdx.x;
  const int wave = tid >> 5, lane = tid & 31;
  const int ml = lane & 15, hi = lane >> 4;
  const int wm = wave & 1, wn = wave >> 1;        // 2 (M) x 4 (N) waves
  const int mBlk = blockIdx.y * GBM, nBlk = blockIdx.x * GBN;

  // staging assignment: A tile 64 rows x 4 x16B chunks (1/thread); B tile 128 x 4 (2/thread)
  const int arow = tid >> 2, achk = tid & 3;
  const size_t  agOff   = (size_t)(mBlk + arow) * K + achk * 8;
  const unsigned aOffL  = (unsigned)(arow * 64 + achk * 16);
  const int brow = tid >> 2, bchk = tid & 3;
  const size_t  bgOff0  = (size_t)(nBlk + brow) * K + bchk * 8;
  const size_t  bgOff1  = (size_t)(nBlk + 64 + brow) * K + bchk * 8;
  const unsigned bOffL0 = (unsigned)(brow * 64 + bchk * 16);
  const unsigned bOffL1 = (unsigned)((64 + brow) * 64 + bchk * 16);
  const unsigned aL[2] = { lds_lo(&Abuf[0][0]), lds_lo(&Abuf[1][0]) };
  const unsigned bL[2] = { lds_lo(&Bbuf[0][0]), lds_lo(&Bbuf[1][0]) };

  auto stage = [&](int buf, int k) {
    async_b128(aL[buf] + aOffL,  A  + agOff  + k);
    async_b128(bL[buf] + bOffL0, BT + bgOff0 + k);
    async_b128(bL[buf] + bOffL1, BT + bgOff1 + k);
  };

  v8f acc00 = {0.f,0.f,0.f,0.f,0.f,0.f,0.f,0.f};
  v8f acc01 = acc00, acc10 = acc00, acc11 = acc00;

  stage(0, 0);
  int buf = 0;
  for (int k = 0; k < K; k += GBK) {
    wait_async0();          // our wave's staged loads for `buf` are in LDS
    __syncthreads();        // everyone's are
    if (k + GBK < K) stage(buf ^ 1, k + GBK);
    const bf16_t* Ab = &Abuf[buf][0];
    const bf16_t* Bb = &Bbuf[buf][0];
    Frag a0, a1, b0, b1;
    const bf16_t* ap0 = Ab + (wm * 32 + ml) * GBK + 16 * hi;
    const bf16_t* ap1 = ap0 + 16 * GBK;
    a0.h[0] = *(const v8bf*)ap0;        a0.h[1] = *(const v8bf*)(ap0 + 8);
    a1.h[0] = *(const v8bf*)ap1;        a1.h[1] = *(const v8bf*)(ap1 + 8);
    const bf16_t* bp0 = Bb + (wn * 32 + ml) * GBK + 16 * hi;
    const bf16_t* bp1 = bp0 + 16 * GBK;
    b0.h[0] = *(const v8bf*)bp0;        b0.h[1] = *(const v8bf*)(bp0 + 8);
    b1.h[0] = *(const v8bf*)bp1;        b1.h[1] = *(const v8bf*)(bp1 + 8);
    acc00 = wmma_bf16(a0.v, b0.v, acc00);
    acc01 = wmma_bf16(a0.v, b1.v, acc01);
    acc10 = wmma_bf16(a1.v, b0.v, acc10);
    acc11 = wmma_bf16(a1.v, b1.v, acc11);
    __syncthreads();        // reads of `buf` done before it is restaged next iter
    buf ^= 1;
  }

  const v8f* accs[2][2] = { { &acc00, &acc01 }, { &acc10, &acc11 } };
  for (int s = 0; s < 2; ++s) {
    for (int t = 0; t < 2; ++t) {
      const v8f& a = *accs[s][t];
      const int n = nBlk + wn * 32 + t * 16 + ml;
      const float bv = bias ? bias[n] : 0.f;
      for (int v = 0; v < 8; ++v) {
        const int m = mBlk + wm * 32 + s * 16 + v + 8 * hi;
        const size_t off = (size_t)z * cStride + (size_t)m * N + n;
        float val = a[v] + bv;
        if (resid) val += resid[off];
        if (Cf) Cf[off] = val;
        if (Cb) Cb[off] = (bf16_t)val;
      }
    }
  }
}

// ---------------- flash attention: 1 wave per 16-row Q tile ----------------
__global__ __launch_bounds__(32) void k_flash(const bf16_t* __restrict__ qkv,
                                              bf16_t* __restrict__ attn) {
  const int qt = blockIdx.x, head = blockIdx.y, b = blockIdx.z;
  const int lane = threadIdx.x & 31;
  const int ml = lane & 15, hi = lane >> 4;
  const size_t rowQ = (size_t)b * S_ + qt * 16;
  const int koff = H_ + head * HD_;
  const int voff = 2 * H_ + head * HD_;
  __shared__ __align__(16) bf16_t pT[16 * 32];   // P tile (16 q x 32 keys)
  __shared__ __align__(16) bf16_t vT[64 * 32];   // V^T tile (64 d x 32 keys)

  Frag a0, a1;   // Q 16x64 as two 16x32 A fragments
  {
    const bf16_t* qp = qkv + (rowQ + ml) * LDQ_ + head * HD_ + 16 * hi;
    a0.h[0] = *(const v8bf*)qp;        a0.h[1] = *(const v8bf*)(qp + 8);
    a1.h[0] = *(const v8bf*)(qp + 32); a1.h[1] = *(const v8bf*)(qp + 40);
  }
  const v8f vzero = {0.f,0.f,0.f,0.f,0.f,0.f,0.f,0.f};
  const float sc = 0.125f;  // HD^-0.5
  float mrow[8], lrow[8];
  v8f o[4];
  for (int v = 0; v < 8; ++v) { mrow[v] = -INFINITY; lrow[v] = 0.f; }
  for (int j = 0; j < 4; ++j) o[j] = vzero;

  const int qmax = qt * 16 + 15;
  for (int kb = 0; kb <= qmax; kb += 32) {
    v8f s0 = vzero, s1 = vzero;
    {
      const bf16_t* kp0 = qkv + ((size_t)b * S_ + kb + ml) * LDQ_ + koff + 16 * hi;
      const bf16_t* kp1 = kp0 + (size_t)16 * LDQ_;
      Frag b00, b01, b10, b11;
      b00.h[0] = *(const v8bf*)kp0;        b00.h[1] = *(const v8bf*)(kp0 + 8);
      b01.h[0] = *(const v8bf*)(kp0 + 32); b01.h[1] = *(const v8bf*)(kp0 + 40);
      b10.h[0] = *(const v8bf*)kp1;        b10.h[1] = *(const v8bf*)(kp1 + 8);
      b11.h[0] = *(const v8bf*)(kp1 + 32); b11.h[1] = *(const v8bf*)(kp1 + 40);
      s0 = wmma_bf16(a0.v, b00.v, s0);  s0 = wmma_bf16(a1.v, b01.v, s0);
      s1 = wmma_bf16(a0.v, b10.v, s1);  s1 = wmma_bf16(a1.v, b11.v, s1);
    }
    float p0[8], p1[8], mx[8], rs[8];
    const int c0 = kb + ml, c1 = c0 + 16;
    for (int v = 0; v < 8; ++v) {
      const int row = qt * 16 + v + 8 * hi;
      p0[v] = (c0 <= row) ? s0[v] * sc : -INFINITY;
      p1[v] = (c1 <= row) ? s1[v] * sc : -INFINITY;
      mx[v] = fmaxf(p0[v], p1[v]);
    }
    for (int off = 1; off < 16; off <<= 1)
      for (int v = 0; v < 8; ++v)
        mx[v] = fmaxf(mx[v], __shfl_xor(mx[v], off, 32));
    for (int v = 0; v < 8; ++v) {
      const float mnew = fmaxf(mrow[v], mx[v]);
      const float alpha = expf(mrow[v] - mnew);
      p0[v] = expf(p0[v] - mnew);
      p1[v] = expf(p1[v] - mnew);
      rs[v] = p0[v] + p1[v];
      lrow[v] *= alpha;
      mrow[v] = mnew;
      for (int j = 0; j < 4; ++j) o[j][v] *= alpha;
    }
    for (int off = 1; off < 16; off <<= 1)
      for (int v = 0; v < 8; ++v)
        rs[v] += __shfl_xor(rs[v], off, 32);
    for (int v = 0; v < 8; ++v) lrow[v] += rs[v];

    for (int v = 0; v < 8; ++v) {
      pT[(v + 8 * hi) * 32 + ml]      = (bf16_t)p0[v];
      pT[(v + 8 * hi) * 32 + ml + 16] = (bf16_t)p1[v];
    }
    {
      const v8bf* vp = (const v8bf*)(qkv + ((size_t)b * S_ + kb + lane) * LDQ_ + voff);
      for (int c = 0; c < 8; ++c) {
        v8bf ch = vp[c];
        for (int e2 = 0; e2 < 8; ++e2)
          vT[(c * 8 + e2) * 32 + lane] = ch[e2];
      }
    }
    __syncthreads();
    Frag pa;
    const bf16_t* pp = pT + ml * 32 + 16 * hi;
    pa.h[0] = *(const v8bf*)pp;  pa.h[1] = *(const v8bf*)(pp + 8);
    for (int j = 0; j < 4; ++j) {
      Frag bv;
      const bf16_t* vp2 = vT + (j * 16 + ml) * 32 + 16 * hi;
      bv.h[0] = *(const v8bf*)vp2;  bv.h[1] = *(const v8bf*)(vp2 + 8);
      o[j] = wmma_bf16(pa.v, bv.v, o[j]);
    }
    __syncthreads();
  }
  for (int v = 0; v < 8; ++v) {
    const float inv = 1.f / lrow[v];
    for (int j = 0; j < 4; ++j)
      attn[(rowQ + v + 8 * hi) * H_ + head * HD_ + j * 16 + ml] =
          (bf16_t)(o[j][v] * inv);
  }
}

// ---------------- router: logits, softmax, lse^2, top-2 (rank-major arrays) ----------------
__global__ __launch_bounds__(256) void k_router(
    const float* __restrict__ xf, const float* __restrict__ rw,
    float* __restrict__ rprobs, float* __restrict__ lse2,
    int* __restrict__ e_rt, float* __restrict__ w_rt) {
  const int t = blockIdx.x, tid = threadIdx.x;
  __shared__ float sh[256][8];
  float acc[8] = {0.f,0.f,0.f,0.f,0.f,0.f,0.f,0.f};
  const float* xr = xf + (size_t)t * H_;
  for (int h = tid; h < H_; h += 256) {
    const float xv = xr[h];
    for (int e = 0; e < 8; ++e) acc[e] += xv * rw[h * E_ + e];
  }
  for (int e = 0; e < 8; ++e) sh[tid][e] = acc[e];
  __syncthreads();
  for (int off = 128; off > 0; off >>= 1) {
    if (tid < off)
      for (int e = 0; e < 8; ++e) sh[tid][e] += sh[tid + off][e];
    __syncthreads();
  }
  if (tid == 0) {
    float lg[8], p[8];
    float mx = sh[0][0];
    for (int e = 0; e < 8; ++e) { lg[e] = sh[0][e]; mx = fmaxf(mx, lg[e]); }
    float se = 0.f;
    for (int e = 0; e < 8; ++e) { p[e] = expf(lg[e] - mx); se += p[e]; }
    const float lse = mx + logf(se);
    lse2[t] = lse * lse;
    const float inv = 1.f / se;
    for (int e = 0; e < 8; ++e) { p[e] *= inv; rprobs[t * E_ + e] = p[e]; }
    int i1 = 0;
    for (int e = 1; e < 8; ++e) if (p[e] > p[i1]) i1 = e;
    int i2 = -1;
    for (int e = 0; e < 8; ++e) {
      if (e == i1) continue;
      if (i2 < 0 || p[e] > p[i2]) i2 = e;
    }
    e_rt[t] = i1;        w_rt[t] = p[i1];
    e_rt[T_ + t] = i2;   w_rt[T_ + t] = p[i2];
  }
}

// ---------------- FIFO queue positions: 8-way prefix scan over 4096 rank-major entries ----------------
__global__ __launch_bounds__(1024) void k_scan(const int* __restrict__ e_rt,
                                               int* __restrict__ pos_rt,
                                               int* __restrict__ kcnt) {
  __shared__ int sh[1024 * 8];
  const int tid = threadIdx.x;
  int loc[8] = {0,0,0,0,0,0,0,0};
  int el[4];
  for (int j = 0; j < 4; ++j) { int e = e_rt[tid * 4 + j]; el[j] = e; loc[e]++; }
  for (int e = 0; e < 8; ++e) sh[tid * 8 + e] = loc[e];
  __syncthreads();
  for (int off = 1; off < 1024; off <<= 1) {
    int v[8] = {0,0,0,0,0,0,0,0};
    if (tid >= off)
      for (int e = 0; e < 8; ++e) v[e] = sh[(tid - off) * 8 + e];
    __syncthreads();
    if (tid >= off)
      for (int e = 0; e < 8; ++e) sh[tid * 8 + e] += v[e];
    __syncthreads();
  }
  int base[8];
  for (int e = 0; e < 8; ++e) base[e] = sh[tid * 8 + e] - loc[e];
  for (int j = 0; j < 4; ++j) { int e = el[j]; pos_rt[tid * 4 + j] = base[e]++; }
  if (tid == 1023)
    for (int e = 0; e < 8; ++e) kcnt[e] = min(sh[1023 * 8 + e], CAP_);
}

// ---------------- dispatch kept tokens into capacity buffers ----------------
__global__ __launch_bounds__(128) void k_dispatch(
    const bf16_t* __restrict__ xf_bf, const int* __restrict__ e_rt,
    const int* __restrict__ pos_rt, bf16_t* __restrict__ xb) {
  const int i = blockIdx.x;          // rank-major entry index
  const int pos = pos_rt[i];
  if (pos >= CAP_) return;
  const int e = e_rt[i], t = i % T_;
  const uint4* src = (const uint4*)(xf_bf + (size_t)t * H_);
  uint4* dst = (uint4*)(xb + ((size_t)e * CAP_ + pos) * H_);
  dst[threadIdx.x] = src[threadIdx.x];   // 128 * 16B = 1024 bf16
}

// ---------------- fused gate/up WMMA (32x32/wave) + SiLU*up -> bf16 ----------------
__global__ __launch_bounds__(256) void k_moe_gateup(
    const bf16_t* __restrict__ XB, const bf16_t* __restrict__ GT,
    const bf16_t* __restrict__ UT, const float* __restrict__ gb,
    const float* __restrict__ ub, bf16_t* __restrict__ Hout) {
  const int e = blockIdx.z;
  const bf16_t* A  = XB + (size_t)e * CAP_ * H_;
  const bf16_t* Gt = GT + (size_t)e * INNER_ * H_;
  const bf16_t* Ut = UT + (size_t)e * INNER_ * H_;
  const int wave = threadIdx.x >> 5, lane = threadIdx.x & 31;
  const int ml = lane & 15, hi = lane >> 4;
  const int wm = wave & 1, wn = wave >> 1;                 // 2 x 4 waves
  const int m0 = blockIdx.y * 64 + wm * 32;
  const int n0 = blockIdx.x * 128 + wn * 32;
  const bf16_t* ap0 = A  + (size_t)(m0 + ml) * H_ + 16 * hi;
  const bf16_t* ap1 = ap0 + (size_t)16 * H_;
  const bf16_t* gp0 = Gt + (size_t)(n0 + ml) * H_ + 16 * hi;
  const bf16_t* gp1 = gp0 + (size_t)16 * H_;
  const bf16_t* up0 = Ut + (size_t)(n0 + ml) * H_ + 16 * hi;
  const bf16_t* up1 = up0 + (size_t)16 * H_;
  v8f g00 = {0.f,0.f,0.f,0.f,0.f,0.f,0.f,0.f};
  v8f g01 = g00, g10 = g00, g11 = g00;
  v8f u00 = g00, u01 = g00, u10 = g00, u11 = g00;
  for (int k = 0; k < H_; k += 32) {
    Frag a0, a1, bg0, bg1, bu0, bu1;
    a0.h[0]  = *(const v8bf*)(ap0 + k);  a0.h[1]  = *(const v8bf*)(ap0 + k + 8);
    a1.h[0]  = *(const v8bf*)(ap1 + k);  a1.h[1]  = *(const v8bf*)(ap1 + k + 8);
    bg0.h[0] = *(const v8bf*)(gp0 + k);  bg0.h[1] = *(const v8bf*)(gp0 + k + 8);
    bg1.h[0] = *(const v8bf*)(gp1 + k);  bg1.h[1] = *(const v8bf*)(gp1 + k + 8);
    bu0.h[0] = *(const v8bf*)(up0 + k);  bu0.h[1] = *(const v8bf*)(up0 + k + 8);
    bu1.h[0] = *(const v8bf*)(up1 + k);  bu1.h[1] = *(const v8bf*)(up1 + k + 8);
    __builtin_prefetch(ap0 + k + 128, 0, 1);
    __builtin_prefetch(gp0 + k + 128, 0, 1);
    __builtin_prefetch(up0 + k + 128, 0, 1);
    g00 = wmma_bf16(a0.v, bg0.v, g00);  g01 = wmma_bf16(a0.v, bg1.v, g01);
    g10 = wmma_bf16(a1.v, bg0.v, g10);  g11 = wmma_bf16(a1.v, bg1.v, g11);
    u00 = wmma_bf16(a0.v, bu0.v, u00);  u01 = wmma_bf16(a0.v, bu1.v, u01);
    u10 = wmma_bf16(a1.v, bu0.v, u10);  u11 = wmma_bf16(a1.v, bu1.v, u11);
  }
  const v8f* ga[2][2] = { { &g00, &g01 }, { &g10, &g11 } };
  const v8f* ua[2][2] = { { &u00, &u01 }, { &u10, &u11 } };
  for (int s = 0; s < 2; ++s) {
    for (int t = 0; t < 2; ++t) {
      const int n = n0 + t * 16 + ml;
      const float gbias = gb[(size_t)e * INNER_ + n];
      const float ubias = ub[(size_t)e * INNER_ + n];
      for (int v = 0; v < 8; ++v) {
        const int m = m0 + s * 16 + v + 8 * hi;
        const float g = (*ga[s][t])[v] + gbias;
        const float u = (*ua[s][t])[v] + ubias;
        const float hsw = g / (1.f + expf(-g)) * u;    // silu(g)*u
        Hout[((size_t)e * CAP_ + m) * INNER_ + n] = (bf16_t)hsw;
      }
    }
  }
}

// ---------------- combine: out = x1 + sum_rank w * y[e,pos] ----------------
__global__ __launch_bounds__(256) void k_combine(
    const float* __restrict__ x1, const float* __restrict__ y,
    const int* __restrict__ e_rt, const int* __restrict__ pos_rt,
    const float* __restrict__ w_rt, float* __restrict__ out) {
  const int t = blockIdx.x;
  int   e0 = e_rt[t],        p0 = pos_rt[t];
  int   e1 = e_rt[T_ + t],   p1 = pos_rt[T_ + t];
  float w0 = w_rt[t],        w1 = w_rt[T_ + t];
  const bool k0 = p0 < CAP_, k1 = p1 < CAP_;
  const float* y0 = y + ((size_t)e0 * CAP_ + (k0 ? p0 : 0)) * H_;
  const float* y1 = y + ((size_t)e1 * CAP_ + (k1 ? p1 : 0)) * H_;
  for (int c = threadIdx.x; c < H_; c += 256) {
    float a = x1[(size_t)t * H_ + c];
    if (k0) a += y0[c] * w0;
    if (k1) a += y1[c] * w1;
    out[(size_t)t * H_ + c] = a;
  }
}

// ---------------- aux / z loss (deterministic tree reductions) ----------------
__global__ __launch_bounds__(256) void k_loss(
    const float* __restrict__ rprobs, const float* __restrict__ lse2,
    const int* __restrict__ kcnt, float* __restrict__ out) {
  __shared__ float shp[256][8];
  __shared__ float shz[256];
  const int tid = threadIdx.x;
  float accp[8] = {0.f,0.f,0.f,0.f,0.f,0.f,0.f,0.f};
  float accz = 0.f;
  for (int t = tid; t < T_; t += 256) {
    for (int e = 0; e < 8; ++e) accp[e] += rprobs[t * E_ + e];
    accz += lse2[t];
  }
  for (int e = 0; e < 8; ++e) shp[tid][e] = accp[e];
  shz[tid] = accz;
  __syncthreads();
  for (int off = 128; off > 0; off >>= 1) {
    if (tid < off) {
      for (int e = 0; e < 8; ++e) shp[tid][e] += shp[tid + off][e];
      shz[tid] += shz[tid + off];
    }
    __syncthreads();
  }
  if (tid == 0) {
    float aux = 0.f;
    for (int e = 0; e < 8; ++e)
      aux += (shp[0][e] / (float)T_) * ((float)kcnt[e] / (float)T_);
    out[0] = aux * (float)E_;
    out[1] = shz[0] / (float)T_;
  }
}

// ---------------- launch ----------------
extern "C" void kernel_launch(void* const* d_in, const int* in_sizes, int n_in,
                              void* d_out, int out_size, void* d_ws, size_t ws_size,
                              hipStream_t stream) {
  const float* x        = (const float*)d_in[0];
  const float* norm1_w  = (const float*)d_in[1];
  const float* norm2_w  = (const float*)d_in[2];
  const float* qkv_w    = (const float*)d_in[3];
  const float* qkv_b    = (const float*)d_in[4];
  const float* out_w    = (const float*)d_in[5];
  const float* out_b    = (const float*)d_in[6];
  const float* router_w = (const float*)d_in[7];
  const float* gate_w   = (const float*)d_in[8];
  const float* gate_b   = (const float*)d_in[9];
  const float* up_w     = (const float*)d_in[10];
  const float* up_b     = (const float*)d_in[11];
  const float* down_w   = (const float*)d_in[12];
  const float* down_b   = (const float*)d_in[13];
  float* out = (float*)d_out;

  char* p = (char*)d_ws;
  auto alloc = [&](size_t bytes) -> char* {
    char* r = p;
    p += (bytes + 255) & ~(size_t)255;
    return r;
  };
  bf16_t* xn_bf   = (bf16_t*)alloc((size_t)T_ * H_ * 2);
  bf16_t* qkvT    = (bf16_t*)alloc((size_t)3 * H_ * H_ * 2);
  bf16_t* qkv_bf  = (bf16_t*)alloc((size_t)T_ * 3 * H_ * 2);
  bf16_t* attn_bf = (bf16_t*)alloc((size_t)T_ * H_ * 2);
  bf16_t* outT    = (bf16_t*)alloc((size_t)H_ * H_ * 2);
  float*  x1      = (float*) alloc((size_t)T_ * H_ * 4);
  bf16_t* xf_bf   = (bf16_t*)alloc((size_t)T_ * H_ * 2);
  float*  xf_f    = (float*) alloc((size_t)T_ * H_ * 4);
  bf16_t* gateT   = (bf16_t*)alloc((size_t)E_ * INNER_ * H_ * 2);
  bf16_t* upT     = (bf16_t*)alloc((size_t)E_ * INNER_ * H_ * 2);
  bf16_t* downT   = (bf16_t*)alloc((size_t)E_ * INNER_ * H_ * 2);
  float*  rprobs  = (float*) alloc((size_t)T_ * E_ * 4);
  float*  lse2    = (float*) alloc((size_t)T_ * 4);
  int*    e_rt    = (int*)   alloc((size_t)2 * T_ * 4);
  int*    pos_rt  = (int*)   alloc((size_t)2 * T_ * 4);
  float*  w_rt    = (float*) alloc((size_t)2 * T_ * 4);
  int*    kcnt    = (int*)   alloc(64);
  bf16_t* xb      = (bf16_t*)alloc((size_t)E_ * CAP_ * H_ * 2);
  bf16_t* h_bf    = (bf16_t*)alloc((size_t)E_ * CAP_ * INNER_ * 2);
  float*  y       = (float*) alloc((size_t)E_ * CAP_ * H_ * 4);

  const dim3 tb(32, 8);
  // weights -> bf16, transposed to [N,K]
  k_transpose_to_bf16<<<dim3(96, 32, 1),   tb, 0, stream>>>(qkv_w,  qkvT,  H_, 3 * H_);
  k_transpose_to_bf16<<<dim3(32, 32, 1),   tb, 0, stream>>>(out_w,  outT,  H_, H_);
  k_transpose_to_bf16<<<dim3(128, 32, E_), tb, 0, stream>>>(gate_w, gateT, H_, INNER_);
  k_transpose_to_bf16<<<dim3(128, 32, E_), tb, 0, stream>>>(up_w,   upT,   H_, INNER_);
  k_transpose_to_bf16<<<dim3(32, 128, E_), tb, 0, stream>>>(down_w, downT, INNER_, H_);

  // attention
  k_rmsnorm<<<T_, 256, 0, stream>>>(x, norm1_w, xn_bf, (float*)nullptr);
  k_gemm_wmma<<<dim3(3 * H_ / GBN, T_ / GBM, 1), 256, 0, stream>>>(
      xn_bf, qkvT, qkv_b, nullptr, nullptr, qkv_bf, T_, 3 * H_, H_, 0, 0, 0, 0);
  k_flash<<<dim3(S_ / 16, NH_, B_), 32, 0, stream>>>(qkv_bf, attn_bf);
  k_gemm_wmma<<<dim3(H_ / GBN, T_ / GBM, 1), 256, 0, stream>>>(
      attn_bf, outT, out_b, x, x1, nullptr, T_, H_, H_, 0, 0, 0, 0);

  // MoE
  k_rmsnorm<<<T_, 256, 0, stream>>>(x1, norm2_w, xf_bf, xf_f);
  k_router<<<T_, 256, 0, stream>>>(xf_f, router_w, rprobs, lse2, e_rt, w_rt);
  k_scan<<<1, 1024, 0, stream>>>(e_rt, pos_rt, kcnt);
  hipMemsetAsync(xb, 0, (size_t)E_ * CAP_ * H_ * 2, stream);
  k_dispatch<<<2 * T_, 128, 0, stream>>>(xf_bf, e_rt, pos_rt, xb);
  k_moe_gateup<<<dim3(INNER_ / 128, CAP_ / 64, E_), 256, 0, stream>>>(
      xb, gateT, upT, gate_b, up_b, h_bf);
  k_gemm_wmma<<<dim3(H_ / GBN, CAP_ / GBM, E_), 256, 0, stream>>>(
      h_bf, downT, down_b, nullptr, y, nullptr, CAP_, H_, INNER_,
      (size_t)CAP_ * INNER_, (size_t)H_ * INNER_, (size_t)CAP_ * H_, (size_t)H_);
  k_combine<<<T_, 256, 0, stream>>>(x1, y, e_rt, pos_rt, w_rt, out);
  k_loss<<<1, 256, 0, stream>>>(rprobs, lse2, kcnt, out + (size_t)T_ * H_);
}
